// CVRP_Decoder_38405597561206
// MI455X (gfx1250) — compile-verified, matrix operand order checked by reading, not run on hardware
//
#include <hip/hip_runtime.h>
#include <hip/hip_bf16.h>

// ---------------------------------------------------------------------------
// CVRP decoder, MI455X (gfx1250, wave32).
// All big einsums use v_wmma_f32_16x16x32_bf16 (bf16 operands, f32 accum):
// bf16 WMMA (16x16x32) gives 8x the flops/instr of f32 WMMA (16x16x4),
// moving the ~32 GFLOP of matmuls under the ~0.8 GB / 23.3 TB/s HBM floor.
// Softmax normalization is fused via a 64 KB LDS probs tile (CDNA5 WGP has
// 320 KB LDS), and phase-2 operand tiles are filled with the CDNA5 async
// copy path (global_load_async_to_lds_b128 + s_wait_asynccnt).
// ---------------------------------------------------------------------------

typedef __attribute__((ext_vector_type(16))) __bf16 v16bf;
typedef __attribute__((ext_vector_type(8)))  float  v8f;

#define Bb 128
#define Nn 512
#define Mm 512
#define Dd 128

// ---- WMMA fragment helpers (ISA 7.12.2 layouts, wave32) -------------------
// A (16x32 bf16) and B-transposed (cols x K, K contiguous) share the same
// per-lane K pattern: lane l: row/col = l&15, kbase = ((l>>4)&1)*8,
// VGPR v holds k = (v>>2)*16 + kbase + (v&3)*2  (+1 for high half).
__device__ __forceinline__ v16bf load_frag(const __bf16* base, int stride, int lane) {
  v16bf f;
  const __bf16* p = base + (size_t)(lane & 15) * stride;
  const int kb = ((lane >> 4) & 1) * 8;
#pragma unroll
  for (int v = 0; v < 8; ++v) {
    int k = ((v >> 2) << 4) + kb + ((v & 3) << 1);
    f[2 * v]     = p[k];
    f[2 * v + 1] = p[k + 1];
  }
  return f;
}

// C/D f32 16x16: lane l -> col = l&15 ; VGPR v -> row = ((l>>4)<<3) + v
__device__ __forceinline__ int c_row(int lane, int v) { return ((lane >> 4) << 3) + v; }

__device__ __forceinline__ v8f wmma_bf16(v16bf a, v16bf b, v8f c) {
  return __builtin_amdgcn_wmma_f32_16x16x32_bf16(
      /*neg_a=*/false, a, /*neg_b=*/false, b,
      /*c_mod=*/(short)0, c, /*reuse_a=*/false, /*reuse_b=*/false);
}

// ---- CDNA5 async copy: global -> LDS, 16B per lane, ASYNCcnt-tracked ------
// GVS addressing: mem = SADDR(sgpr64) + VADDR(vgpr u32) ; VDST = LDS byte addr.
__device__ __forceinline__ void async_copy_b128(unsigned lds_addr, const void* sbase,
                                                unsigned voff) {
  asm volatile("global_load_async_to_lds_b128 %0, %1, %2"
               :: "v"(lds_addr), "v"(voff), "s"(sbase)
               : "memory");
}
__device__ __forceinline__ void wait_async0() {
  asm volatile("s_wait_asynccnt 0" ::: "memory");
}

// ---------------------------------------------------------------------------
// Kernel 1: k = E@Wk^T, v = E@Wv^T ; store exp(k), exp(k)*v transposed
// as bf16 [b][e][m] (direct B-fragment feed for phase 1), and a bf16 copy
// of encoded_nodes [b][m][e] for phase-2 async LDS fills.
// grid = B * (M/128), block = 256 (8 waves; wave w owns row-tile w).
// ---------------------------------------------------------------------------
__global__ __launch_bounds__(256)
void kv_kernel(const float* __restrict__ enc, const float* __restrict__ Wk,
               const float* __restrict__ Wv, __bf16* __restrict__ ws_ek,
               __bf16* __restrict__ ws_ekv, __bf16* __restrict__ ws_encb) {
  __shared__ __bf16 sE[128 * 128];  // 32 KB
  __shared__ __bf16 sW[128 * 128];  // 32 KB
  const int b  = blockIdx.x >> 2;
  const int mt = (blockIdx.x & 3) << 7;
  const int tid = threadIdx.x, wave = tid >> 5, lane = tid & 31;
  const size_t tile0 = ((size_t)b * Mm + mt) * Dd;

#pragma unroll 4
  for (int i = 0; i < 64; ++i) {
    int idx = tid + (i << 8);
    __bf16 e = (__bf16)enc[tile0 + idx];
    sE[idx] = e;
    ws_encb[tile0 + idx] = e;           // bf16 copy for phase-2 async fills
    sW[idx] = (__bf16)Wk[idx];          // Wk row-major == B-transposed [e][d]
  }
  __syncthreads();

  v8f ck[8], cv[8];
  v8f z = {};
#pragma unroll
  for (int t = 0; t < 8; ++t) ck[t] = z;
#pragma unroll
  for (int kc = 0; kc < 4; ++kc) {
    v16bf a = load_frag(sE + (wave << 4) * 128 + (kc << 5), 128, lane);
    v16bf bcur = load_frag(sW + (kc << 5), 128, lane);
#pragma unroll
    for (int ct = 0; ct < 8; ++ct) {
      v16bf bnext = bcur;
      if (ct < 7) bnext = load_frag(sW + ((ct + 1) << 4) * 128 + (kc << 5), 128, lane);
      ck[ct] = wmma_bf16(a, bcur, ck[ct]);
      bcur = bnext;
    }
  }
  __syncthreads();
#pragma unroll 4
  for (int i = 0; i < 64; ++i) {
    int idx = tid + (i << 8);
    sW[idx] = (__bf16)Wv[idx];
  }
  __syncthreads();
#pragma unroll
  for (int t = 0; t < 8; ++t) cv[t] = z;
#pragma unroll
  for (int kc = 0; kc < 4; ++kc) {
    v16bf a = load_frag(sE + (wave << 4) * 128 + (kc << 5), 128, lane);
    v16bf bcur = load_frag(sW + (kc << 5), 128, lane);
#pragma unroll
    for (int ct = 0; ct < 8; ++ct) {
      v16bf bnext = bcur;
      if (ct < 7) bnext = load_frag(sW + ((ct + 1) << 4) * 128 + (kc << 5), 128, lane);
      cv[ct] = wmma_bf16(a, bcur, cv[ct]);
      bcur = bnext;
    }
  }
  // elementwise: exp(k), exp(k)*v ; store transposed [b][e][m]
#pragma unroll
  for (int ct = 0; ct < 8; ++ct) {
#pragma unroll
    for (int v = 0; v < 8; ++v) {
      int e = (ct << 4) + (lane & 15);
      int m = mt + (wave << 4) + c_row(lane, v);
      float ek  = expf(ck[ct][v]);
      float ekv = ek * cv[ct][v];
      size_t o = ((size_t)b * Dd + e) * Mm + m;
      ws_ek[o]  = (__bf16)ek;
      ws_ekv[o] = (__bf16)ekv;
    }
  }
}

// ---------------------------------------------------------------------------
// Kernel 2: q = [E_last, load] @ Wq^T ; store sigmoid(q) f32 [b][n][e].
// grid = B * (N/128), block = 256.
// ---------------------------------------------------------------------------
__global__ __launch_bounds__(256)
void q_kernel(const float* __restrict__ eln, const float* __restrict__ ld,
              const float* __restrict__ Wq, float* __restrict__ ws_sigq) {
  __shared__ __bf16 sA[128 * 128];
  __shared__ __bf16 sW[128 * 128];
  const int b  = blockIdx.x >> 2;
  const int nb = (blockIdx.x & 3) << 7;
  const int tid = threadIdx.x, wave = tid >> 5, lane = tid & 31;

#pragma unroll 4
  for (int i = 0; i < 64; ++i) {
    int idx = tid + (i << 8);
    sA[idx] = (__bf16)eln[((size_t)b * Nn + nb + (idx >> 7)) * Dd + (idx & 127)];
    sW[idx] = (__bf16)Wq[(size_t)(idx >> 7) * 129 + (idx & 127)];
  }
  __syncthreads();

  v8f c[8];
  v8f z = {};
#pragma unroll
  for (int t = 0; t < 8; ++t) c[t] = z;
#pragma unroll
  for (int kc = 0; kc < 4; ++kc) {
    v16bf a = load_frag(sA + (wave << 4) * 128 + (kc << 5), 128, lane);
    v16bf bcur = load_frag(sW + (kc << 5), 128, lane);
#pragma unroll
    for (int ct = 0; ct < 8; ++ct) {
      v16bf bnext = bcur;
      if (ct < 7) bnext = load_frag(sW + ((ct + 1) << 4) * 128 + (kc << 5), 128, lane);
      c[ct] = wmma_bf16(a, bcur, c[ct]);
      bcur = bnext;
    }
  }
#pragma unroll
  for (int ct = 0; ct < 8; ++ct) {
#pragma unroll
    for (int v = 0; v < 8; ++v) {
      int e = (ct << 4) + (lane & 15);
      int n = nb + (wave << 4) + c_row(lane, v);
      float q = c[ct][v] + ld[(size_t)b * Nn + n] * Wq[(size_t)e * 129 + 128];
      ws_sigq[((size_t)b * Nn + n) * Dd + e] = 1.f / (1.f + expf(-q));
    }
  }
}

// ---------------------------------------------------------------------------
// Kernel 3: fused AAFM + decode score + tanh-clip + FULL softmax (probs tile
// held in LDS; single write of the final output).
// grid = B * (N/64), block = 256 (8 waves; wave w owns 16-col tile w).
// Static LDS = 16 + 32 + 64 + 0.25 KB = ~112.5 KB (< 320 KB WGP LDS).
// ---------------------------------------------------------------------------
__global__ __launch_bounds__(256)
void attn_kernel(const float* __restrict__ cur_dist, const float* __restrict__ ninf,
                 const float* __restrict__ log_scale, const float* __restrict__ alpha1,
                 const float* __restrict__ alpha2, const __bf16* __restrict__ ws_ek,
                 const __bf16* __restrict__ ws_ekv, const float* __restrict__ ws_sigq,
                 const __bf16* __restrict__ ws_encb, float* __restrict__ out) {
  __shared__ __bf16 sAAFM[64 * 128];   // 16 KB (phase-2 A matrix)
  __shared__ __bf16 sScr[128 * 128];   // 32 KB (phase-1 eb / phase-2 enc)
  __shared__ __bf16 sP[64 * 512];      // 64 KB (unnormalized probs)
  __shared__ float  sRow[64];
  const int b  = blockIdx.x >> 3;
  const int nb = (blockIdx.x & 7) << 6;
  const int tid = threadIdx.x, wave = tid >> 5, lane = tid & 31;
  const float ls = log_scale[0];
  const float a1 = alpha1[0] * ls, a2 = alpha2[0] * ls;
  if (tid < 64) sRow[tid] = 0.f;

  v8f z = {};
  v8f num[4], den[4];
#pragma unroll
  for (int rt = 0; rt < 4; ++rt) { num[rt] = z; den[rt] = z; }

  // ---- Phase 1: num/den = eb @ (e^k * v), eb @ e^k -------------------------
  for (int mb = 0; mb < 4; ++mb) {
#pragma unroll 4
    for (int i = 0; i < 32; ++i) {  // fill eb tile (64 x 128) as bf16
      int idx = tid + (i << 8);
      int nl = idx >> 7, mc = idx & 127;
      size_t gi = ((size_t)b * Nn + nb + nl) * Mm + (mb << 7) + mc;
      sScr[idx] = (__bf16)expf(-a1 * cur_dist[gi] + ninf[gi]);
    }
    __syncthreads();
#pragma unroll
    for (int kc = 0; kc < 4; ++kc) {
      int kg = (mb << 7) + (kc << 5);
      const __bf16* bkbase = ws_ek  + ((size_t)b * Dd + (wave << 4)) * Mm + kg;
      const __bf16* bvbase = ws_ekv + ((size_t)b * Dd + (wave << 4)) * Mm + kg;
      v16bf bek = load_frag(bkbase, Mm, lane);
      v16bf bev = load_frag(bvbase, Mm, lane);
      v16bf acur = load_frag(sScr + (kc << 5), 128, lane);
#pragma unroll
      for (int rt = 0; rt < 4; ++rt) {
        v16bf anext = acur;
        if (rt < 3) anext = load_frag(sScr + ((rt + 1) << 4) * 128 + (kc << 5), 128, lane);
        num[rt] = wmma_bf16(acur, bev, num[rt]);
        den[rt] = wmma_bf16(acur, bek, den[rt]);
        acur = anext;
      }
    }
    __syncthreads();
  }

  // aafm = sigmoid(q) * nan_to_num(num/den) -> LDS bf16 (A for phase 2)
#pragma unroll
  for (int rt = 0; rt < 4; ++rt) {
#pragma unroll
    for (int v = 0; v < 8; ++v) {
      int nl = (rt << 4) + c_row(lane, v);
      int e  = (wave << 4) + (lane & 15);
      float dd = den[rt][v];
      float w  = (dd != 0.f) ? num[rt][v] / dd : 0.f;
      float sg = ws_sigq[((size_t)b * Nn + nb + nl) * Dd + e];
      sAAFM[nl * 128 + e] = (__bf16)(sg * w);
    }
  }
  __syncthreads();

  // ---- Phase 2: score = aafm @ enc^T ; p = exp(10*tanh(.) + mask) ----------
  const float inv_sqrt_d = 0.08838834764831845f;  // 1/sqrt(128)
  const unsigned sScr_lds = (unsigned)(uintptr_t)(void*)sScr;
  for (int mb = 0; mb < 4; ++mb) {
    // enc bf16 tile (128 x 128, [m][e]) via CDNA5 async copy: 16B per lane.
    const __bf16* encb_tile = ws_encb + ((size_t)b * Mm + (mb << 7)) * Dd;
#pragma unroll
    for (int i = 0; i < 8; ++i) {
      unsigned idx16 = (unsigned)tid + (i << 8);     // 2048 x 16B = 32 KB
      async_copy_b128(sScr_lds + idx16 * 16u, encb_tile, idx16 * 16u);
    }
    wait_async0();
    __syncthreads();

    v8f sc[4];
#pragma unroll
    for (int rt = 0; rt < 4; ++rt) sc[rt] = z;
#pragma unroll
    for (int kc = 0; kc < 4; ++kc) {
      v16bf bf = load_frag(sScr + (wave << 4) * 128 + (kc << 5), 128, lane);
      v16bf acur = load_frag(sAAFM + (kc << 5), 128, lane);
#pragma unroll
      for (int rt = 0; rt < 4; ++rt) {
        v16bf anext = acur;
        if (rt < 3) anext = load_frag(sAAFM + ((rt + 1) << 4) * 128 + (kc << 5), 128, lane);
        sc[rt] = wmma_bf16(acur, bf, sc[rt]);
        acur = anext;
      }
    }
#pragma unroll
    for (int rt = 0; rt < 4; ++rt) {
#pragma unroll
      for (int v = 0; v < 8; ++v) {
        int nl = (rt << 4) + c_row(lane, v);
        int mc = (mb << 7) + (wave << 4) + (lane & 15);
        size_t gi = ((size_t)b * Nn + nb + nl) * Mm + mc;
        float sv = sc[rt][v] * inv_sqrt_d - a2 * cur_dist[gi];
        float p  = expf(10.f * tanhf(sv) + ninf[gi]);  // bounded, no max-sub needed
        sP[nl * 512 + mc] = (__bf16)p;
        // reduce p over the 16 lanes sharing this row
        float r = p;
        r += __shfl_xor(r, 1);
        r += __shfl_xor(r, 2);
        r += __shfl_xor(r, 4);
        r += __shfl_xor(r, 8);
        if ((lane & 15) == 0) atomicAdd(&sRow[nl], r);
      }
    }
    __syncthreads();
  }

  // fused normalization: probs = p / rowsum, single float4 store per 4 elems
  __syncthreads();
#pragma unroll 4
  for (int i = 0; i < 32; ++i) {
    int idx4 = tid + (i << 8);          // 8192 float4 in the 64x512 tile
    int nl = idx4 >> 7;                 // 128 float4 per row
    int mc = (idx4 & 127) << 2;
    float inv = 1.f / sRow[nl];
    float4 p;
    p.x = (float)sP[nl * 512 + mc + 0] * inv;
    p.y = (float)sP[nl * 512 + mc + 1] * inv;
    p.z = (float)sP[nl * 512 + mc + 2] * inv;
    p.w = (float)sP[nl * 512 + mc + 3] * inv;
    *((float4*)(out + ((size_t)b * Nn + nb + nl) * Mm + mc)) = p;
  }
}

// ---------------------------------------------------------------------------
extern "C" void kernel_launch(void* const* d_in, const int* in_sizes, int n_in,
                              void* d_out, int out_size, void* d_ws, size_t ws_size,
                              hipStream_t stream) {
  const float* eln       = (const float*)d_in[0];   // (B,N,D)
  const float* ld        = (const float*)d_in[1];   // (B,N)
  const float* cur_dist  = (const float*)d_in[2];   // (B,N,M)
  const float* log_scale = (const float*)d_in[3];   // ()
  const float* ninf      = (const float*)d_in[4];   // (B,N,M)
  const float* enc       = (const float*)d_in[5];   // (B,M,D)
  const float* Wq        = (const float*)d_in[6];   // (D,D+1)
  const float* Wk        = (const float*)d_in[7];   // (D,D)
  const float* Wv        = (const float*)d_in[8];   // (D,D)
  const float* alpha1    = (const float*)d_in[9];   // (1,)
  const float* alpha2    = (const float*)d_in[10];  // (1,)
  float* out = (float*)d_out;

  // workspace layout (~80.5 MB)
  char* ws = (char*)d_ws;
  const size_t n_ekd = (size_t)Bb * Dd * Mm;              // 8,388,608 elems
  __bf16* ws_ek   = (__bf16*)ws;                          // 16 MB
  __bf16* ws_ekv  = (__bf16*)(ws + n_ekd * 2);            // 16 MB
  float*  ws_sigq = (float*)(ws + n_ekd * 4);             // 32 MB
  __bf16* ws_encb = (__bf16*)(ws + n_ekd * 4 + (size_t)Bb * Nn * Dd * 4);  // 16 MB

  kv_kernel<<<Bb * (Mm / 128), 256, 0, stream>>>(enc, Wk, Wv, ws_ek, ws_ekv, ws_encb);
  q_kernel<<<Bb * (Nn / 128), 256, 0, stream>>>(eln, ld, Wq, ws_sigq);
  attn_kernel<<<Bb * (Nn / 64), 256, 0, stream>>>(
      cur_dist, ninf, log_scale, alpha1, alpha2,
      ws_ek, ws_ekv, ws_sigq, ws_encb, out);
}
